// WaterfallTransformerLayer3D_16879221473446
// MI455X (gfx1250) — matrix operand order repaired and checked by reading, not compiled, not run on hardware
//
#include <hip/hip_runtime.h>
#include <hip/hip_bf16.h>

typedef __attribute__((ext_vector_type(16))) _Float16 v16h;
typedef __attribute__((ext_vector_type(8)))  float    v8f;
typedef __attribute__((ext_vector_type(4)))  unsigned int u32x4;
typedef __attribute__((ext_vector_type(8)))  int      i32x8;
typedef __attribute__((ext_vector_type(4)))  int      i32x4;

// ---- problem dims ----
constexpr int Cc   = 96;     // channels
constexpr int NTOK = 128;    // tokens per window (2*8*8)
constexpr int NWIN = 1536;   // B * nd * nh * nw = 8*4*8*6
// LDS strides in halves (multiples of 8 halves = 16B for ds_load_b128 alignment)
constexpr int YST = 104;     // f16 stride, 96-wide token rows
constexpr int TST = 136;     // f16 stride, 128-wide Vt rows (channel-major V)
constexpr int SST = 132;     // f32 stride, 128-wide score rows
constexpr int PST = 136;     // f16 stride, 128-wide prob rows
constexpr int HST = 392;     // f16 stride, 384-wide hidden rows
constexpr int XST = 100;     // f32 stride, 96-wide residual rows

// window/token -> flat token index in (B,D,H,W) order
__device__ __forceinline__ long long tok2flat(int w, int t) {
    int b  = w / 192;  int r = w % 192;        // nd*nh*nw = 4*8*6
    int d0 = r / 48;   r %= 48;                // nh*nw = 48
    int h0 = r / 6;    int w0 = r % 6;
    int wd = t >> 6;   int wh = (t >> 3) & 7;  int ww = t & 7;
    int Dg = d0 * 2 + wd, Hg = h0 * 8 + wh, Wg = w0 * 8 + ww;
    return (((long long)b * 8 + Dg) * 64 + Hg) * 48 + Wg;
}

// A-fragment (16x32 f16) from LDS, token-major layout.
__device__ __forceinline__ v16h load_a(const _Float16* base, int stride, int mt, int lane) {
    const _Float16* row = base + (mt * 16 + (lane & 15)) * stride + ((lane >> 4) << 4);
    v16h a;
#pragma unroll
    for (int i = 0; i < 8; ++i) a[i] = row[i];
#pragma unroll
    for (int i = 0; i < 8; ++i) a[8 + i] = row[8 + i];
    return a;
}

#define WMMA_F16(A, B, C) \
    __builtin_amdgcn_wmma_f32_16x16x32_f16(false, (A), false, (B), (short)0, (C), false, false)

// Issue one TDM 3D-tile load: window (2 x 8 x 768 f32) from x -> LDS
__device__ __forceinline__ void tdm_load_window(const float* gsrc, void* ldst) {
    unsigned long long ga  = (unsigned long long)(size_t)gsrc;
    unsigned           lds = (unsigned)(size_t)ldst;
    u32x4 g0;
    g0[0] = 1u;                                          // count=1, no gather
    g0[1] = lds;                                         // lds_addr (bytes)
    g0[2] = (unsigned)(ga & 0xFFFFFFFFu);                // global_addr[31:0]
    g0[3] = (unsigned)((ga >> 32) & 0x01FFFFFFu) | (2u << 30);  // addr[56:32] | type=2
    i32x8 g1;
    g1[0] = (int)(2u << 16);                             // data_size = 4B
    g1[1] = (int)((4608u & 0xFFFFu) << 16);              // tensor_dim0 lo16 (=W*C row)
    g1[2] = (int)((4608u >> 16) | (64u << 16));          // td0 hi16 | tensor_dim1 lo16
    g1[3] = (int)(768u << 16);                           // td1 hi16=0 | tile_dim0=768
    g1[4] = (int)(8u | (2u << 16));                      // tile_dim1=8, tile_dim2=2
    g1[5] = (int)4608;                                   // tensor_dim0_stride lo32
    g1[6] = (int)((294912u & 0xFFFFu) << 16);            // st0 hi16=0 | st1 lo16
    g1[7] = (int)(294912u >> 16);                        // tensor_dim1_stride[47:16]
    i32x4 g2; g2[0] = (int)(1u << 20); g2[1] = 0; g2[2] = 0; g2[3] = 0;
    i32x4 g3; g3[0] = 0; g3[1] = 0; g3[2] = 0; g3[3] = 0;
#if defined(__clang_major__) && (__clang_major__ >= 23)
    i32x8 g4; g4[0]=0; g4[1]=0; g4[2]=0; g4[3]=0; g4[4]=0; g4[5]=0; g4[6]=0; g4[7]=0;
    __builtin_amdgcn_tensor_load_to_lds(g0, g1, g2, g3, g4, 0);
#else
    __builtin_amdgcn_tensor_load_to_lds(g0, g1, g2, g3, 0);
#endif
}

// ---- pack weight (Nout x Kin, row-major) into WMMA B-fragment order, f16 ----
__global__ void pack_b_kernel(const float* __restrict__ W, _Float16* __restrict__ out,
                              int K, int N) {
    int idx = blockIdx.x * 256 + threadIdx.x;
    int KC = K >> 5;
    int total = (N >> 4) * KC * 512;
    if (idx >= total) return;
    int i    = idx & 15;
    int lane = (idx >> 4) & 31;
    int rest = idx >> 9;
    int kc   = rest % KC;
    int nt   = rest / KC;
    int n = nt * 16 + (lane & 15);
    int k = kc * 32 + ((lane >> 4) << 4) + i;
    out[idx] = (_Float16)W[n * K + k];
}

// ---- kernel A: LN1 + window attention + proj + residual -> out (=x1) ----
__global__ __launch_bounds__(256)
void attn_kernel(const float* __restrict__ x,
                 const float* __restrict__ g1, const float* __restrict__ b1,
                 const _Float16* __restrict__ qkvP, const float* __restrict__ qkv_b,
                 const _Float16* __restrict__ projP, const float* __restrict__ proj_b,
                 float* __restrict__ out) {
    extern __shared__ char smem[];
    _Float16* Y  = (_Float16*)smem;             // LN output, later attn output O
    _Float16* Q  = Y + NTOK * YST;
    _Float16* K  = Q + NTOK * YST;
    _Float16* Vt = K + NTOK * YST;              // V transposed: [channel][token]
    float*    S  = (float*)(Vt + Cc * TST);     // scores (one head at a time)
    _Float16* P  = (_Float16*)(S + NTOK * SST);
    float*    XS = (float*)(P + NTOK * PST);    // TDM-staged x window (token-major)

    const int w    = blockIdx.x;
    const int tid  = threadIdx.x;
    const int wave = tid >> 5;                  // == M-tile for all GEMMs
    const int lane = tid & 31;

    __builtin_prefetch(qkvP, 0, 0);
    __builtin_prefetch(projP, 0, 0);

    // ---- phase 0: TDM DMA of the whole window into LDS (one instruction) ----
    if (wave == 0) {
        tdm_load_window(x + tok2flat(w, 0) * Cc, XS);
        __builtin_amdgcn_s_wait_tensorcnt(0);
    }
    __syncthreads();

    // ---- phase 1: LayerNorm from staged window -> Y (f16) ----
    if (tid < NTOK) {
        const float4* xr = (const float4*)(XS + tid * Cc);
        float buf[Cc];
        float m = 0.f;
#pragma unroll
        for (int i = 0; i < Cc / 4; ++i) {
            float4 v4 = xr[i];
            buf[4 * i + 0] = v4.x; buf[4 * i + 1] = v4.y;
            buf[4 * i + 2] = v4.z; buf[4 * i + 3] = v4.w;
            m += v4.x + v4.y + v4.z + v4.w;
        }
        m *= (1.f / Cc);
        float var = 0.f;
#pragma unroll
        for (int c = 0; c < Cc; ++c) { float d = buf[c] - m; var += d * d; }
        float rs = rsqrtf(var * (1.f / Cc) + 1e-5f);
#pragma unroll
        for (int c = 0; c < Cc; ++c)
            Y[tid * YST + c] = (_Float16)((buf[c] - m) * rs * g1[c] + b1[c]);
    }
    __syncthreads();

    // ---- phase 2: QKV = Y @ qkv_w^T + b ; wave = M-tile, sweep 18 N-tiles ----
    {
        v16h aF[3];
#pragma unroll
        for (int kc = 0; kc < 3; ++kc) aF[kc] = load_a(Y + kc * 32, YST, wave, lane);
        for (int nt = 0; nt < 18; ++nt) {
            v8f acc = {};
#pragma unroll
            for (int kc = 0; kc < 3; ++kc) {
                v16h b = *(const v16h*)(qkvP + (((nt * 3 + kc) << 5) + lane) * 16);
                acc = WMMA_F16(aF[kc], b, acc);
            }
            int n = nt * 16 + (lane & 15);
            float bias = qkv_b[n];
#pragma unroll
            for (int r = 0; r < 8; ++r) {
                int mrow = wave * 16 + r + ((lane >> 4) << 3);
                float v = acc[r] + bias;
                if (nt < 6) {          // Q, scaled by hd^-1/2
                    Q[mrow * YST + n] = (_Float16)(v * 0.1767766952966369f);
                } else if (nt < 12) {  // K
                    K[mrow * YST + (n - 96)] = (_Float16)v;
                } else {               // V, stored transposed
                    Vt[(n - 192) * TST + mrow] = (_Float16)v;
                }
            }
        }
    }
    __syncthreads();

    // ---- phase 3: per-head attention; O written into Y ----
    for (int h = 0; h < 3; ++h) {
        // scores S = Q_h @ K_h^T (head K dim = 32 -> one WMMA); wave = M-tile
        {
            v16h a = load_a(Q + h * 32, YST, wave, lane);
            for (int nt = 0; nt < 8; ++nt) {
                v16h b;
                const _Float16* kp = K + (nt * 16 + (lane & 15)) * YST + h * 32
                                       + ((lane >> 4) << 4);
#pragma unroll
                for (int i = 0; i < 16; ++i) b[i] = kp[i];
                v8f acc = {};
                acc = WMMA_F16(a, b, acc);
                int n = nt * 16 + (lane & 15);
#pragma unroll
                for (int r = 0; r < 8; ++r)
                    S[(wave * 16 + r + ((lane >> 4) << 3)) * SST + n] = acc[r];
            }
        }
        __syncthreads();
        // softmax rows -> P (f16)
        if (tid < NTOK) {
            float* sr = S + tid * SST;
            float mx = -1e30f;
            for (int j = 0; j < NTOK; ++j) mx = fmaxf(mx, sr[j]);
            float sum = 0.f;
            for (int j = 0; j < NTOK; ++j) sum += __expf(sr[j] - mx);
            float inv = 1.f / sum;
            for (int j = 0; j < NTOK; ++j)
                P[tid * PST + j] = (_Float16)(__expf(sr[j] - mx) * inv);
        }
        __syncthreads();
        // O_h = P @ V_h  (K=128 -> 4 chunks; N=32 -> 2 tiles); wave = M-tile
        {
            v8f acc0 = {}, acc1 = {};
            for (int kc = 0; kc < 4; ++kc) {
                v16h a = load_a(P + kc * 32, PST, wave, lane);
                const _Float16* vp0 = Vt + (h * 32 + (lane & 15)) * TST
                                         + kc * 32 + ((lane >> 4) << 4);
                v16h b0, b1v;
#pragma unroll
                for (int i = 0; i < 16; ++i) b0[i] = vp0[i];
                const _Float16* vp1 = vp0 + 16 * TST;
#pragma unroll
                for (int i = 0; i < 16; ++i) b1v[i] = vp1[i];
                acc0 = WMMA_F16(a, b0, acc0);
                acc1 = WMMA_F16(a, b1v, acc1);
            }
            int col = h * 32 + (lane & 15);
#pragma unroll
            for (int r = 0; r < 8; ++r) {
                int mrow = wave * 16 + r + ((lane >> 4) << 3);
                Y[mrow * YST + col]      = (_Float16)acc0[r];
                Y[mrow * YST + col + 16] = (_Float16)acc1[r];
            }
        }
        __syncthreads();
    }

    // ---- phase 4: proj + residual -> out ; wave = M-tile, 6 N-tiles in regs ----
    {
        v16h aF[3];
#pragma unroll
        for (int kc = 0; kc < 3; ++kc) aF[kc] = load_a(Y + kc * 32, YST, wave, lane);
        v8f acc[6] = {};
        for (int kc = 0; kc < 3; ++kc) {
#pragma unroll
            for (int nt = 0; nt < 6; ++nt) {
                v16h b = *(const v16h*)(projP + (((nt * 3 + kc) << 5) + lane) * 16);
                acc[nt] = WMMA_F16(aF[kc], b, acc[nt]);
            }
        }
#pragma unroll
        for (int nt = 0; nt < 6; ++nt) {
            int n = nt * 16 + (lane & 15);
            float bias = proj_b[n];
#pragma unroll
            for (int r = 0; r < 8; ++r) {
                int t = wave * 16 + r + ((lane >> 4) << 3);
                long long g = tok2flat(w, t) * Cc + n;
                out[g] = XS[t * Cc + n] + acc[nt][r] + bias;   // residual from staged x
            }
        }
    }
}

// ---- kernel B: LN2 + MLP(GELU exact) + residual, in-place on out ----
__global__ __launch_bounds__(256)
void mlp_kernel(const float* __restrict__ g2, const float* __restrict__ b2,
                const _Float16* __restrict__ fc1P, const float* __restrict__ fc1_b,
                const _Float16* __restrict__ fc2P, const float* __restrict__ fc2_b,
                float* __restrict__ out) {
    extern __shared__ char smem[];
    float*    X1 = (float*)smem;                   // residual rows f32
    _Float16* Hn = (_Float16*)(X1 + NTOK * XST);   // LN output f16
    _Float16* Hd = Hn + NTOK * YST;                // hidden f16

    const int tid  = threadIdx.x;
    const int wave = tid >> 5;                     // == M-tile
    const int lane = tid & 31;
    const long long rowbase = (long long)blockIdx.x * NTOK;

    __builtin_prefetch(fc1P, 0, 0);
    __builtin_prefetch(fc2P, 0, 0);

    if (tid < NTOK) {
        const float4* xr = (const float4*)(out + (rowbase + tid) * Cc);
        float buf[Cc];
        float m = 0.f;
#pragma unroll
        for (int i = 0; i < Cc / 4; ++i) {
            float4 v4 = xr[i];
            buf[4 * i + 0] = v4.x; buf[4 * i + 1] = v4.y;
            buf[4 * i + 2] = v4.z; buf[4 * i + 3] = v4.w;
            m += v4.x + v4.y + v4.z + v4.w;
        }
        m *= (1.f / Cc);
        float var = 0.f;
#pragma unroll
        for (int c = 0; c < Cc; ++c) { float d = buf[c] - m; var += d * d; }
        float rs = rsqrtf(var * (1.f / Cc) + 1e-5f);
#pragma unroll
        for (int c = 0; c < Cc; ++c) {
            X1[tid * XST + c] = buf[c];
            Hn[tid * YST + c] = (_Float16)((buf[c] - m) * rs * g2[c] + b2[c]);
        }
    }
    __syncthreads();

    // fc1 + exact GELU -> Hd ; wave = M-tile, sweep 24 N-tiles, A in regs
    {
        v16h aF[3];
#pragma unroll
        for (int kc = 0; kc < 3; ++kc) aF[kc] = load_a(Hn + kc * 32, YST, wave, lane);
        for (int nt = 0; nt < 24; ++nt) {
            v8f acc = {};
#pragma unroll
            for (int kc = 0; kc < 3; ++kc) {
                v16h b = *(const v16h*)(fc1P + (((nt * 3 + kc) << 5) + lane) * 16);
                acc = WMMA_F16(aF[kc], b, acc);
            }
            int n = nt * 16 + (lane & 15);
            float bias = fc1_b[n];
#pragma unroll
            for (int r = 0; r < 8; ++r) {
                int t = wave * 16 + r + ((lane >> 4) << 3);
                float v = acc[r] + bias;
                float ge = 0.5f * v * (1.f + erff(v * 0.7071067811865475f));
                Hd[t * HST + n] = (_Float16)ge;
            }
        }
    }
    __syncthreads();

    // fc2 + residual -> out ; wave = M-tile, 6 N-tile accumulators, K outer
    {
        v8f acc[6] = {};
        for (int kc = 0; kc < 12; ++kc) {
            v16h a = load_a(Hd + kc * 32, HST, wave, lane);
#pragma unroll
            for (int nt = 0; nt < 6; ++nt) {
                v16h b = *(const v16h*)(fc2P + (((nt * 12 + kc) << 5) + lane) * 16);
                acc[nt] = WMMA_F16(a, b, acc[nt]);
            }
        }
#pragma unroll
        for (int nt = 0; nt < 6; ++nt) {
            int n = nt * 16 + (lane & 15);
            float bias = fc2_b[n];
#pragma unroll
            for (int r = 0; r < 8; ++r) {
                int t = wave * 16 + r + ((lane >> 4) << 3);
                out[(rowbase + t) * Cc + n] = X1[t * XST + n] + acc[nt][r] + bias;
            }
        }
    }
}

extern "C" void kernel_launch(void* const* d_in, const int* in_sizes, int n_in,
                              void* d_out, int out_size, void* d_ws, size_t ws_size,
                              hipStream_t stream) {
    const float* x      = (const float*)d_in[0];
    const float* g1     = (const float*)d_in[1];
    const float* b1     = (const float*)d_in[2];
    const float* qkv_w  = (const float*)d_in[3];
    const float* qkv_b  = (const float*)d_in[4];
    const float* proj_w = (const float*)d_in[5];
    const float* proj_b = (const float*)d_in[6];
    const float* g2     = (const float*)d_in[7];
    const float* b2     = (const float*)d_in[8];
    const float* fc1_w  = (const float*)d_in[9];
    const float* fc1_b  = (const float*)d_in[10];
    const float* fc2_w  = (const float*)d_in[11];
    const float* fc2_b  = (const float*)d_in[12];
    float* out = (float*)d_out;

    // workspace carve: packed f16 B-fragment weights
    _Float16* qkvP  = (_Float16*)d_ws;       // 18*3*512 = 27648 halves
    _Float16* projP = qkvP + 27648;          //  6*3*512 =  9216
    _Float16* fc1P  = projP + 9216;          // 24*3*512 = 36864
    _Float16* fc2P  = fc1P + 36864;          //  6*12*512= 36864

    pack_b_kernel<<<(27648 + 255) / 256, 256, 0, stream>>>(qkv_w,  qkvP,  96,  288);
    pack_b_kernel<<<( 9216 + 255) / 256, 256, 0, stream>>>(proj_w, projP, 96,  96);
    pack_b_kernel<<<(36864 + 255) / 256, 256, 0, stream>>>(fc1_w,  fc1P,  96,  384);
    pack_b_kernel<<<(36864 + 255) / 256, 256, 0, stream>>>(fc2_w,  fc2P,  384, 96);

    size_t smemA = (size_t)(3 * NTOK * YST) * 2 + (size_t)(Cc * TST) * 2
                 + (size_t)(NTOK * SST) * 4 + (size_t)(NTOK * PST) * 2
                 + (size_t)(NTOK * Cc) * 4;                               // 257536 B
    attn_kernel<<<NWIN, 256, smemA, stream>>>(x, g1, b1, qkvP, qkv_b, projP, proj_b, out);

    size_t smemB = (size_t)(NTOK * XST) * 4 + (size_t)(NTOK * YST) * 2
                 + (size_t)(NTOK * HST) * 2;                              // 178176 B
    mlp_kernel<<<NWIN, 256, smemB, stream>>>(g2, b2, fc1P, fc1_b, fc2P, fc2_b, out);
}
